// VASNet_62654982914288
// MI455X (gfx1250) — matrix operand order repaired and checked by reading, not compiled, not run on hardware
//
#include <hip/hip_runtime.h>

typedef __attribute__((ext_vector_type(16))) __bf16 v16bf;
typedef __attribute__((ext_vector_type(8)))  __bf16 v8bf;
typedef __attribute__((ext_vector_type(8)))  float  v8f;

#define N_ROWS 32768
#define MDIM   1024

// quaternion component table: index [ci*4 + co]
__constant__ int   d_widx[16] = {0,1,2,3,  1,0,3,2,  2,3,0,1,  3,2,1,0};
__constant__ float d_sgn[16]  = {1.f,1.f,1.f,1.f,  -1.f,1.f,-1.f,1.f,
                                 -1.f,1.f,1.f,-1.f, -1.f,-1.f,1.f,1.f};

__device__ inline __bf16 f2bf(float x) {
  union { float f; unsigned u; } in; in.f = x;
  unsigned r = in.u + 0x7FFFu + ((in.u >> 16) & 1u);   // round-to-nearest-even
  union { unsigned short s; __bf16 h; } out;
  out.s = (unsigned short)(r >> 16);
  return out.h;
}

// Expand quaternion weight (4,256,256) f32 -> transposed bf16 Wbig^T[n][k] (1024x1024)
__global__ __launch_bounds__(256) void build_wbig(const float* __restrict__ W,
                                                  __bf16* __restrict__ Bt) {
  int idx = blockIdx.x * 256 + threadIdx.x;           // 1M elements
  int nn = idx >> 10, kk = idx & 1023;
  int co = nn >> 8, o = nn & 255, ci = kk >> 8, i = kk & 255;
  int t = ci * 4 + co;
  float v = d_sgn[t] * W[d_widx[t] * 65536 + i * 256 + o];
  Bt[(size_t)nn * 1024 + kk] = f2bf(v);
}

// bulk f32 -> bf16 (8 elements/thread)
__global__ __launch_bounds__(256) void cvt_bf16(const float* __restrict__ X,
                                                __bf16* __restrict__ Y) {
  size_t i0 = ((size_t)blockIdx.x * 256 + threadIdx.x) * 8;
  const float4* p = (const float4*)(X + i0);
  float4 a = p[0], b = p[1];
  v8bf o;
  o[0]=f2bf(a.x); o[1]=f2bf(a.y); o[2]=f2bf(a.z); o[3]=f2bf(a.w);
  o[4]=f2bf(b.x); o[5]=f2bf(b.y); o[6]=f2bf(b.z); o[7]=f2bf(b.w);
  *(v8bf*)(Y + i0) = o;
}

// C(MxN) = A(MxK,bf16) * Bt(NxK,bf16)^T, f32 accumulate; C = acc*scale (+ bias)
// 256 threads = 8 waves (4Mx2N); block tile 128x128; wave tile 32x64 = 2x4 wmma;
// BK=32; double-buffered LDS, one barrier per K-step. Template kills epilogue
// branches; min 2 waves/SIMD keeps VGPR cap high enough to avoid acc spills.
template <bool BF_OUT, bool HAS_BIAS>
__global__ __launch_bounds__(256, 2) void gemm_bf16(const __bf16* __restrict__ A,
                                                    const __bf16* __restrict__ Bt,
                                                    float* Cf, __bf16* Cb,
                                                    const float* bias, float scale,
                                                    int M, int N, int K) {
  __shared__ __bf16 lA[2][128][48];   // 32 data + 16 pad; 96B stride (16B aligned)
  __shared__ __bf16 lB[2][128][48];

  const int tid  = threadIdx.x;
  const int lane = tid & 31, wave = tid >> 5;
  const int wm = wave >> 1, wn = wave & 1;            // 4 x 2 wave grid
  const int l16 = lane & 15, lh = lane >> 4;
  const int m0 = blockIdx.y * 128, n0 = blockIdx.x * 128;
  const int r = tid >> 1, s = tid & 1;                // stager: 128 rows x 32B

  v8f acc[2][4];
  for (int i = 0; i < 2; ++i)
    for (int j = 0; j < 4; ++j)
      for (int e = 0; e < 8; ++e) acc[i][j][e] = 0.f;

  auto stage = [&](int pb, int k0) {
    const v8bf* ap = (const v8bf*)(A + (size_t)(m0 + r) * K + k0 + s * 16);
    *(v8bf*)&lA[pb][r][s * 16]     = ap[0];
    *(v8bf*)&lA[pb][r][s * 16 + 8] = ap[1];
    const v8bf* bp = (const v8bf*)(Bt + (size_t)(n0 + r) * K + k0 + s * 16);
    *(v8bf*)&lB[pb][r][s * 16]     = bp[0];
    *(v8bf*)&lB[pb][r][s * 16 + 8] = bp[1];
  };

  stage(0, 0);
  __syncthreads();
  int p = 0;
  for (int k0 = 0; k0 < K; k0 += 32) {
    v16bf af[2], bfm[4];
    for (int sm = 0; sm < 2; ++sm) {
      int arow = wm * 32 + sm * 16 + l16;
      v8bf alo = *(const v8bf*)&lA[p][arow][lh * 8];        // K 0..7 / 8..15
      v8bf ahi = *(const v8bf*)&lA[p][arow][16 + lh * 8];   // K 16..23 / 24..31
      af[sm] = __builtin_shufflevector(alo, ahi, 0,1,2,3,4,5,6,7,
                                                 8,9,10,11,12,13,14,15);
    }
    for (int sn = 0; sn < 4; ++sn) {
      int bcol = wn * 64 + sn * 16 + l16;
      v8bf blo = *(const v8bf*)&lB[p][bcol][lh * 8];        // K 0..7 / 8..15
      v8bf bhi = *(const v8bf*)&lB[p][bcol][16 + lh * 8];   // K 16..23 / 24..31
      bfm[sn] = __builtin_shufflevector(blo, bhi, 0,1,2,3,4,5,6,7,
                                                  8,9,10,11,12,13,14,15);
    }
    if (k0 + 32 < K) stage(p ^ 1, k0 + 32);
    __syncthreads();
    for (int sm = 0; sm < 2; ++sm)
      for (int sn = 0; sn < 4; ++sn)
        acc[sm][sn] = __builtin_amdgcn_wmma_f32_16x16x32_bf16(
            false, af[sm], false, bfm[sn], (short)0, acc[sm][sn], false, false);
    p ^= 1;
  }

  for (int sm = 0; sm < 2; ++sm)
    for (int sn = 0; sn < 4; ++sn) {
      const int col  = n0 + wn * 64 + sn * 16 + l16;
      const int rbase = m0 + wm * 32 + sm * 16 + lh * 8;    // VGPR e -> M=e / e+8
      float badd = HAS_BIAS ? bias[col] : 0.f;
      if (BF_OUT) {
        __bf16* cp = Cb + (size_t)rbase * N + col;
        for (int e = 0; e < 8; ++e)
          cp[(size_t)e * N] = f2bf(acc[sm][sn][e] * scale + badd);
      } else {
        float* cp = Cf + (size_t)rbase * N + col;
        for (int e = 0; e < 8; ++e)
          cp[(size_t)e * N] = acc[sm][sn][e] * scale + badd;
      }
    }
}

// logits = Q (x) K (Hamilton), att = q_normalize(logits) -> f32 out, y1 = att (x) V -> bf16
// Kb and y1 may alias (each element read before written by the same thread).
__global__ __launch_bounds__(256) void attn_elem(const __bf16* Qb, const __bf16* Kb,
                                                 const __bf16* Vb, float* att,
                                                 __bf16* y1) {
  int idx = blockIdx.x * 256 + threadIdx.x;            // n*256
  int row = idx >> 8, d = idx & 255;
  size_t b0 = (size_t)row * 1024 + d;
  float a = (float)Qb[b0], b = (float)Qb[b0+256];
  float c = (float)Qb[b0+512], dd = (float)Qb[b0+768];
  float e = (float)Kb[b0], f = (float)Kb[b0+256];
  float g = (float)Kb[b0+512], h = (float)Kb[b0+768];
  float r  = a*e - b*f - c*g - dd*h;
  float i_ = a*f + b*e + c*h - dd*g;
  float j_ = a*g - b*h + c*e + dd*f;
  float k_ = a*h + b*g - c*f + dd*e;
  float nrm = sqrtf(r*r + i_*i_ + j_*j_ + k_*k_) + 1e-4f;
  float ar = r/nrm, ai = i_/nrm, aj = j_/nrm, ak = k_/nrm;
  att[b0] = ar; att[b0+256] = ai; att[b0+512] = aj; att[b0+768] = ak;
  float ve = (float)Vb[b0], vf = (float)Vb[b0+256];
  float vg = (float)Vb[b0+512], vh = (float)Vb[b0+768];
  y1[b0]     = f2bf(ar*ve - ai*vf - aj*vg - ak*vh);
  y1[b0+256] = f2bf(ar*vf + ai*ve + aj*vh - ak*vg);
  y1[b0+512] = f2bf(ar*vg - ai*vh + aj*ve + ak*vf);
  y1[b0+768] = f2bf(ar*vh + ai*vg - aj*vf + ak*ve);
}

// out = g * (x - mean)/(std_unbiased + 1e-6) + b ; x = X (+R), optional pre-relu.
// outF may alias X (all row reads precede writes). outF/outB each nullable.
__global__ __launch_bounds__(256) void rowln(const float* X, const float* R,
                                             const float* __restrict__ g,
                                             const float* __restrict__ b,
                                             float* outF, __bf16* outB,
                                             int relu_flag) {
  int row = blockIdx.x, t = threadIdx.x;
  const float* xr = X + (size_t)row * 1024;
  const float* rr = R ? R + (size_t)row * 1024 : nullptr;
  float v[4], s = 0.f, ss = 0.f;
  for (int i = 0; i < 4; ++i) {
    int c = t + i * 256;
    float x = xr[c];
    if (rr) x += rr[c];
    if (relu_flag) x = fmaxf(x, 0.f);
    v[i] = x; s += x; ss += x * x;
  }
  __shared__ float sh1[256], sh2[256];
  sh1[t] = s; sh2[t] = ss; __syncthreads();
  for (int o = 128; o > 0; o >>= 1) {
    if (t < o) { sh1[t] += sh1[t+o]; sh2[t] += sh2[t+o]; }
    __syncthreads();
  }
  float mean = sh1[0] * (1.f / 1024.f);
  float var  = (sh2[0] - 1024.f * mean * mean) * (1.f / 1023.f);
  float inv  = 1.f / (sqrtf(fmaxf(var, 0.f)) + 1e-6f);
  for (int i = 0; i < 4; ++i) {
    int c = t + i * 256;
    float y = g[c] * (v[i] - mean) * inv + b[c];
    if (outF) outF[(size_t)row * 1024 + c] = y;
    if (outB) outB[(size_t)row * 1024 + c] = f2bf(y);
  }
}

// y6 = sigmoid(qlinear(y5, Wkd(4,256,1), bkd)) -> out[row*4 + c]
__global__ __launch_bounds__(256) void final_head(const float* __restrict__ y5,
                                                  const float* __restrict__ Wkd,
                                                  const float* __restrict__ bkd,
                                                  float* __restrict__ out) {
  int row = blockIdx.x, t = threadIdx.x;
  const float* yr = y5 + (size_t)row * 1024;
  __shared__ float sh[16][256];
  for (int ci = 0; ci < 4; ++ci) {
    float v = yr[ci * 256 + t];
    for (int cw = 0; cw < 4; ++cw)
      sh[ci * 4 + cw][t] = v * Wkd[cw * 256 + t];
  }
  __syncthreads();
  for (int o = 128; o > 0; o >>= 1) {
    if (t < o)
      for (int q = 0; q < 16; ++q) sh[q][t] += sh[q][t + o];
    __syncthreads();
  }
  if (t == 0) {
    float P[16];
    for (int q = 0; q < 16; ++q) P[q] = sh[q][0];
    float o0 = P[0]  - P[5]  - P[10] - P[15] + bkd[0];  // r
    float o1 = P[1]  + P[4]  + P[11] - P[14] + bkd[1];  // i
    float o2 = P[2]  - P[7]  + P[8]  + P[13] + bkd[2];  // j
    float o3 = P[3]  + P[6]  - P[9]  + P[12] + bkd[3];  // k
    out[(size_t)row * 4 + 0] = 1.f / (1.f + expf(-o0));
    out[(size_t)row * 4 + 1] = 1.f / (1.f + expf(-o1));
    out[(size_t)row * 4 + 2] = 1.f / (1.f + expf(-o2));
    out[(size_t)row * 4 + 3] = 1.f / (1.f + expf(-o3));
  }
}

extern "C" void kernel_launch(void* const* d_in, const int* in_sizes, int n_in,
                              void* d_out, int out_size, void* d_ws, size_t ws_size,
                              hipStream_t stream) {
  (void)in_sizes; (void)n_in; (void)out_size; (void)ws_size;
  const int n = N_ROWS;
  const float* x    = (const float*)d_in[0];
  const float* WK   = (const float*)d_in[1];
  const float* WQ   = (const float*)d_in[2];
  const float* WV   = (const float*)d_in[3];
  const float* Wout = (const float*)d_in[4];
  const float* g_y  = (const float*)d_in[5];
  const float* b_y  = (const float*)d_in[6];
  const float* Wka  = (const float*)d_in[7];
  const float* bka  = (const float*)d_in[8];
  const float* g_ka = (const float*)d_in[9];
  const float* b_ka2= (const float*)d_in[10];
  const float* Wkd  = (const float*)d_in[11];
  const float* bkd  = (const float*)d_in[12];

  char* ws = (char*)d_ws;
  const size_t WB  = (size_t)MDIM * MDIM * sizeof(__bf16);   // 2MB per Wbig
  const size_t ACT = (size_t)n * MDIM * sizeof(__bf16);      // 64MB bf16 activation
  __bf16* wbK = (__bf16*)(ws + 0 * WB);
  __bf16* wbQ = (__bf16*)(ws + 1 * WB);
  __bf16* wbV = (__bf16*)(ws + 2 * WB);
  __bf16* wbO = (__bf16*)(ws + 3 * WB);
  __bf16* wbA = (__bf16*)(ws + 4 * WB);
  char* base = ws + 5 * WB;
  __bf16* xbf  = (__bf16*)(base + 0 * ACT);
  __bf16* Kb   = (__bf16*)(base + 1 * ACT);   // K, later y1
  __bf16* Qb   = (__bf16*)(base + 2 * ACT);   // Q, later y3
  __bf16* Vb   = (__bf16*)(base + 3 * ACT);
  float*  buf1 = (float*)(base + 4 * ACT);    // 128MB f32: y2, then y4/y5 in place

  float* out_y   = (float*)d_out;             // (n,4) flattened
  float* out_att = out_y + (size_t)4 * n;     // (n,1024)

  const int wgrid = (MDIM * MDIM) / 256;      // 4096
  build_wbig<<<wgrid, 256, 0, stream>>>(WK,   wbK);
  build_wbig<<<wgrid, 256, 0, stream>>>(WQ,   wbQ);
  build_wbig<<<wgrid, 256, 0, stream>>>(WV,   wbV);
  build_wbig<<<wgrid, 256, 0, stream>>>(Wout, wbO);
  build_wbig<<<wgrid, 256, 0, stream>>>(Wka,  wbA);
  cvt_bf16<<<(n * MDIM) / (256 * 8), 256, 0, stream>>>(x, xbf);

  dim3 gg(MDIM / 128, n / 128);               // (8, 256)
  gemm_bf16<true,  false><<<gg, 256, 0, stream>>>(xbf, wbK, nullptr, Kb, nullptr, 1.0f,  n, MDIM, MDIM);
  gemm_bf16<true,  false><<<gg, 256, 0, stream>>>(xbf, wbQ, nullptr, Qb, nullptr, 0.06f, n, MDIM, MDIM);
  gemm_bf16<true,  false><<<gg, 256, 0, stream>>>(xbf, wbV, nullptr, Vb, nullptr, 1.0f,  n, MDIM, MDIM);

  attn_elem<<<n, 256, 0, stream>>>(Qb, Kb, Vb, out_att, Kb);          // y1 over K

  gemm_bf16<false, false><<<gg, 256, 0, stream>>>(Kb, wbO, buf1, nullptr, nullptr, 1.0f, n, MDIM, MDIM); // y2
  rowln<<<n, 256, 0, stream>>>(buf1, x, g_y, b_y, nullptr, Qb, 0);    // y3 (bf16) over Q
  gemm_bf16<false, true ><<<gg, 256, 0, stream>>>(Qb, wbA, buf1, nullptr, bka, 1.0f, n, MDIM, MDIM);     // y4
  rowln<<<n, 256, 0, stream>>>(buf1, nullptr, g_ka, b_ka2, buf1, nullptr, 1);                            // y5
  final_head<<<n, 256, 0, stream>>>(buf1, Wkd, bkd, out_y);
}
